// SelfAttention_65206193488304
// MI455X (gfx1250) — matrix operand order repaired and checked by reading, not compiled, not run on hardware
//
#include <hip/hip_runtime.h>
#include <math.h>

typedef _Float16 half_t;
typedef _Float16 v16h __attribute__((ext_vector_type(16)));
typedef _Float16 v8h  __attribute__((ext_vector_type(8)));
typedef float    v8f  __attribute__((ext_vector_type(8)));
typedef int      v4i_ __attribute__((vector_size(16)));

#define B_   2
#define T_   1024
#define D_   1024
#define H_   16
#define DH_  64
#define KV_  3072
#define SCALE_ 0.125f

#if __has_builtin(__builtin_amdgcn_global_load_async_to_lds_b128) && \
    __has_builtin(__builtin_amdgcn_s_wait_asynccnt)
#define ASYNC_LDS 1
#endif

// Build a v16h WMMA operand from two contiguous 8-half runs in LDS.
// A-operand: runs at (base + 16*hi) and (base + 16*hi + 8)  [K = e+16hi]
// B-operand: runs at (base + 16*hi) and (base + 16*hi + 8)  [K = e+16hi]
__device__ __forceinline__ v16h ld_op(const half_t* p0, const half_t* p1) {
  v8h lo = *(const v8h*)p0;
  v8h hi = *(const v8h*)p1;
  return __builtin_shufflevector(lo, hi, 0,1,2,3,4,5,6,7,8,9,10,11,12,13,14,15);
}

__device__ __forceinline__ void cp16(half_t* ldst, const half_t* gsrc) {
#ifdef ASYNC_LDS
  __builtin_amdgcn_global_load_async_to_lds_b128(
      (__attribute__((address_space(1))) v4i_*)(void*)gsrc,
      (__attribute__((address_space(3))) v4i_*)(void*)ldst, 0, 0);
#else
  *(float4*)ldst = *(const float4*)gsrc;
#endif
}
__device__ __forceinline__ void cp_wait() {
#ifdef ASYNC_LDS
  __builtin_amdgcn_s_wait_asynccnt(0);
#endif
}

// ---------------------------------------------------------------------------
// WMMA GEMM: C(MxN) = alpha * A(MxK) @ B(KxN) (+bias). B passed TRANSPOSED
// (BT is NxK row-major) so both operands gather as contiguous b128 runs.
// Block = 256 threads = 8 waves; block tile 64x128; wave tile 16x64.
// M%64==0, N%128==0, K%32==0.
// ---------------------------------------------------------------------------
__global__ __launch_bounds__(256) void gemm_wmma(
    const half_t* __restrict__ A, const half_t* __restrict__ BT,
    int M, int N, int K, int lda, int ldbt, int ldc,
    float alpha, const float* __restrict__ bias,
    float* __restrict__ Cf, half_t* __restrict__ Ch)
{
  __shared__ __align__(16) half_t sA [64 * 32];
  __shared__ __align__(16) half_t sBT[128 * 32];
  const int tid  = threadIdx.x;
  const int warp = tid >> 5;
  const int lane = tid & 31;
  const int hi   = lane >> 4;
  const int ln   = lane & 15;
  const int m0   = blockIdx.y * 64;
  const int n0b  = blockIdx.x * 128;
  const int rb   = (warp & 3) * 16;
  const int cb   = (warp >> 2) * 64;

  v8f acc[4] = {v8f{}, v8f{}, v8f{}, v8f{}};
  for (int k0 = 0; k0 < K; k0 += 32) {
    // A tile 64x32 = 256 b128 (1/thread); BT tile 128x32 = 512 b128 (2/thread)
    {
      int row = tid >> 2, off = tid & 3;
      cp16(sA + tid * 8, A + (size_t)(m0 + row) * lda + k0 + off * 8);
    }
    #pragma unroll
    for (int i = 0; i < 2; ++i) {
      int f = tid * 2 + i;
      int row = f >> 2, off = f & 3;
      cp16(sBT + f * 8, BT + (size_t)(n0b + row) * ldbt + k0 + off * 8);
    }
    cp_wait();
    __syncthreads();
    const half_t* ab = sA + (rb + ln) * 32;
    v16h a = ld_op(ab + 16 * hi, ab + 16 * hi + 8);
    #pragma unroll
    for (int t = 0; t < 4; ++t) {
      const half_t* bb = sBT + (cb + t * 16 + ln) * 32 + 16 * hi;
      acc[t] = __builtin_amdgcn_wmma_f32_16x16x32_f16(
          false, a, false, ld_op(bb, bb + 8), (short)0, acc[t], false, false);
    }
    __syncthreads();
  }
  #pragma unroll
  for (int t = 0; t < 4; ++t)
    #pragma unroll
    for (int r = 0; r < 8; ++r) {
      int row = m0 + rb + r + 8 * hi;
      int col = n0b + cb + t * 16 + ln;
      float v = alpha * acc[t][r];
      if (bias) v += bias[col];
      if (Cf) Cf[(size_t)row * ldc + col] = v;
      if (Ch) Ch[(size_t)row * ldc + col] = (half_t)v;
    }
}

// ---------------------------------------------------------------------------
// Flash attention with WMMA-computed relative-position skew term.
// q pre-scaled by SCALE. One wave per 16-row query tile, 32-key steps.
// All LDS wave-private -> no block barriers needed. 2 waves / block.
// ---------------------------------------------------------------------------
#define ATT_WAVES 2
#define WAVE_LDS  8704   // Q 1024 | K 2048 | V 2048 | P 512 | Pos 48*64
__global__ __launch_bounds__(64) void attn_kernel(
    const half_t* __restrict__ qh, const half_t* __restrict__ kvh,
    const half_t* __restrict__ posh, half_t* __restrict__ outh)
{
  __shared__ __align__(16) half_t lds[ATT_WAVES * WAVE_LDS];
  const int tid  = threadIdx.x;
  const int warp = tid >> 5;
  const int lane = tid & 31;
  const int hi   = lane >> 4;
  const int ln   = lane & 15;
  half_t* sQ   = lds + warp * WAVE_LDS;
  half_t* sK   = sQ + 1024;
  half_t* sV   = sK + 2048;
  half_t* sP   = sV + 2048;
  half_t* sPos = sP + 512;

  const int w  = blockIdx.x * ATT_WAVES + warp;   // 0..2047
  const int qt = w & 63;
  const int h  = (w >> 6) & 15;
  const int b  = w >> 10;
  const int r0 = qt * 16;

  // Q tile 16x64 : 128 b128, 4 per lane
  #pragma unroll
  for (int i = 0; i < 4; ++i) {
    int f = lane * 4 + i;
    int row = f >> 3, off = f & 7;
    cp16(sQ + f * 8, qh + (size_t)(b * T_ + r0 + row) * D_ + h * DH_ + off * 8);
  }

  float m_[8], l_[8], sc[8];
  v8f acc[4] = {v8f{}, v8f{}, v8f{}, v8f{}};
  #pragma unroll
  for (int r = 0; r < 8; ++r) { m_[r] = -INFINITY; l_[r] = 0.f; }

  for (int k0 = 0; k0 < KV_; k0 += 32) {
    const int j0 = 1023 + k0 - r0;
    // K,V tiles 32x64 (8 b128/lane each); pos band 48x64 (12 b128/lane)
    #pragma unroll
    for (int i = 0; i < 8; ++i) {
      int f = lane * 8 + i;
      int row = f >> 3, off = f & 7;
      size_t base = (size_t)(b * KV_ + k0 + row) * 2048;
      cp16(sK + f * 8, kvh + base + h * DH_ + off * 8);
      cp16(sV + f * 8, kvh + base + 1024 + h * DH_ + off * 8);
    }
    #pragma unroll
    for (int i = 0; i < 12; ++i) {
      int f = lane * 12 + i;
      int row = f >> 3, off = f & 7;
      int j = j0 - 15 + row; if (j > KV_ - 1) j = KV_ - 1;
      cp16(sPos + f * 8, posh + ((size_t)h * KV_ + j) * DH_ + off * 8);
    }
    if (k0 + 32 < KV_)
      __builtin_prefetch(kvh + (size_t)(b * KV_ + k0 + 32) * 2048 + h * DH_, 0, 1);
    cp_wait();

    // S = Q@K^T (4 WMMA) and band M[g] = Q@Pband^T (6 WMMA), shared A operand
    v8f s2[2] = {v8f{}, v8f{}};
    v8f M[3]  = {v8f{}, v8f{}, v8f{}};
    #pragma unroll
    for (int dsl = 0; dsl < 2; ++dsl) {
      const half_t* qb = sQ + ln * 64 + dsl * 32;
      v16h a = ld_op(qb + 16 * hi, qb + 16 * hi + 8);
      #pragma unroll
      for (int g = 0; g < 2; ++g) {
        const half_t* kb = sK + (g * 16 + ln) * 64 + dsl * 32 + 16 * hi;
        s2[g] = __builtin_amdgcn_wmma_f32_16x16x32_f16(
            false, a, false, ld_op(kb, kb + 8), (short)0, s2[g], false, false);
      }
      #pragma unroll
      for (int g = 0; g < 3; ++g) {
        const half_t* pb = sPos + (g * 16 + ln) * 64 + dsl * 32 + 16 * hi;
        M[g] = __builtin_amdgcn_wmma_f32_16x16x32_f16(
            false, a, false, ld_op(pb, pb + 8), (short)0, M[g], false, false);
      }
    }

    // extract skewed diagonal: pos(rc,kc) = M[.][rc, 15+kc-rc]
    #pragma unroll
    for (int gS = 0; gS < 2; ++gS)
      #pragma unroll
      for (int r = 0; r < 8; ++r) {
        int idx = 15 + gS * 16 + ln - r - 8 * hi;     // 0..46
        int src = hi * 16 + (idx & 15);
        int g   = idx >> 4;
        float v0 = __shfl(M[0][r], src, 32);
        float v1 = __shfl(M[1][r], src, 32);
        float v2 = __shfl(M[2][r], src, 32);
        float pv = (g == 0) ? v0 : ((g == 1) ? v1 : v2);
        if (j0 - 15 + idx <= KV_ - 1) s2[gS][r] += pv;
      }

    // online softmax
    #pragma unroll
    for (int r = 0; r < 8; ++r) {
      float rm = fmaxf(s2[0][r], s2[1][r]);
      rm = fmaxf(rm, __shfl_xor(rm, 1, 16));
      rm = fmaxf(rm, __shfl_xor(rm, 2, 16));
      rm = fmaxf(rm, __shfl_xor(rm, 4, 16));
      rm = fmaxf(rm, __shfl_xor(rm, 8, 16));
      float mn = fmaxf(m_[r], rm);
      sc[r] = __expf(m_[r] - mn);
      float p0 = __expf(s2[0][r] - mn);
      float p1 = __expf(s2[1][r] - mn);
      float rs = p0 + p1;
      rs += __shfl_xor(rs, 1, 16);
      rs += __shfl_xor(rs, 2, 16);
      rs += __shfl_xor(rs, 4, 16);
      rs += __shfl_xor(rs, 8, 16);
      l_[r] = l_[r] * sc[r] + rs;
      m_[r] = mn;
      int rc = r + 8 * hi;
      sP[rc * 32 + ln]      = (half_t)p0;
      sP[rc * 32 + 16 + ln] = (half_t)p1;
    }
    #pragma unroll
    for (int t = 0; t < 4; ++t)
      #pragma unroll
      for (int r = 0; r < 8; ++r)
        acc[t][r] *= sc[r];

    // acc += P(16x32) @ V(32x64)
    const half_t* pbse = sP + ln * 32;
    v16h pa = ld_op(pbse + 16 * hi, pbse + 16 * hi + 8);
    #pragma unroll
    for (int t = 0; t < 4; ++t) {
      v16h vb;
      #pragma unroll
      for (int e = 0; e < 16; ++e)
        vb[e] = sV[(e + 16 * hi) * 64 + t * 16 + ln];
      acc[t] = __builtin_amdgcn_wmma_f32_16x16x32_f16(
          false, pa, false, vb, (short)0, acc[t], false, false);
    }
  }

  #pragma unroll
  for (int t = 0; t < 4; ++t)
    #pragma unroll
    for (int r = 0; r < 8; ++r) {
      int rc = r + 8 * hi;
      outh[(size_t)(b * T_ + r0 + rc) * D_ + h * DH_ + t * 16 + ln] =
          (half_t)(acc[t][r] / l_[r]);
    }
}

// ---------------------------------------------------------------------------
// Pack / convert / assemble kernels
// ---------------------------------------------------------------------------
__global__ void cvt_f32_f16(half_t* dst, const float* src, int n) {
  int i = blockIdx.x * 256 + threadIdx.x;
  if (i < n) dst[i] = (half_t)src[i];
}

// src (RxC) f32 -> dst (CxR) f16
__global__ void cvt_transpose(half_t* dst, const float* src, int R, int C) {
  int i = blockIdx.x * 256 + threadIdx.x;
  if (i >= R * C) return;
  int r = i / C, c = i % C;
  dst[(size_t)c * R + r] = (half_t)src[i];
}

__global__ void build_kvin(half_t* dst, const float* cmem, const float* mem,
                           const float* x) {
  int i = blockIdx.x * 256 + threadIdx.x;           // B*KV*D
  if (i >= B_ * KV_ * D_) return;
  int col = i & 1023;
  int row = i >> 10;
  int b = row / KV_;
  int s = row % KV_;
  float v;
  if (s < 1024)       v = cmem[((size_t)b * 1024 + s) * 1024 + col];
  else if (s < 2048)  v = mem [((size_t)b * 1024 + (s - 1024)) * 1024 + col];
  else                v = x   [((size_t)b * 1024 + (s - 2048)) * 1024 + col];
  dst[i] = (half_t)v;
}

// W2T[o, r*1024+i] = conv_w[o, i, r]   (transposed conv weight, 1024x4096)
__global__ void build_convWT(half_t* dst, const float* cw) {
  int i = blockIdx.x * 256 + threadIdx.x;           // 1024*4096
  if (i >= 1024 * 4096) return;
  int o  = i >> 12;
  int c  = i & 4095;
  int r  = c >> 10;
  int ii = c & 1023;
  dst[i] = (half_t)cw[(size_t)o * 4096 + ii * 4 + r];
}

__global__ void copy_f32(float* dst, const float* src, int n) {
  int i = blockIdx.x * 256 + threadIdx.x;
  if (i < n) dst[i] = src[i];
}

__global__ void build_cmem(float* dst, const float* cmem, const float* comp) {
  int i = blockIdx.x * 256 + threadIdx.x;           // B*1024*1024
  if (i >= B_ * 1024 * 1024) return;
  int col = i & 1023;
  int rr  = i >> 10;
  int b   = rr >> 10;
  int row = rr & 1023;
  dst[i] = (row < 768)
      ? cmem[((size_t)b * 1024 + 256 + row) * 1024 + col]
      : comp[((size_t)(b * 256 + row - 768)) * 1024 + col];
}

__global__ void write_aux(float* dst) { if (threadIdx.x == 0) dst[0] = 0.f; }

// ---------------------------------------------------------------------------
extern "C" void kernel_launch(void* const* d_in, const int* in_sizes, int n_in,
                              void* d_out, int out_size, void* d_ws, size_t ws_size,
                              hipStream_t stream) {
  const float* x      = (const float*)d_in[0];
  const float* mem    = (const float*)d_in[1];
  const float* cmem   = (const float*)d_in[2];
  const float* pos    = (const float*)d_in[3];
  const float* Wq     = (const float*)d_in[4];
  const float* Wkv    = (const float*)d_in[5];
  const float* Wout   = (const float*)d_in[6];
  const float* b_out  = (const float*)d_in[7];
  const float* conv_w = (const float*)d_in[8];
  const float* conv_b = (const float*)d_in[9];
  float* out = (float*)d_out;

  char* ws = (char*)d_ws;
  size_t cur = 0;
  auto alloc = [&](size_t bytes) { char* p = ws + cur; cur += (bytes + 255) & ~size_t(255); return p; };
  half_t* xh     = (half_t*)alloc((size_t)2048 * 1024 * 2);
  half_t* wqt    = (half_t*)alloc((size_t)1024 * 1024 * 2);   // Wq^T  1024x1024
  half_t* wkvt   = (half_t*)alloc((size_t)2048 * 1024 * 2);   // Wkv^T 2048x1024
  half_t* woutt  = (half_t*)alloc((size_t)1024 * 1024 * 2);   // Wout^T
  half_t* kvinh  = (half_t*)alloc((size_t)6144 * 1024 * 2);
  half_t* qh     = (half_t*)alloc((size_t)2048 * 1024 * 2);
  half_t* kvh    = (half_t*)alloc((size_t)6144 * 2048 * 2);
  half_t* outh   = (half_t*)alloc((size_t)2048 * 1024 * 2);
  half_t* convAh = (half_t*)alloc((size_t)512 * 4096 * 2);
  half_t* convWT = (half_t*)alloc((size_t)1024 * 4096 * 2);
  half_t* posh   = (half_t*)alloc((size_t)H_ * KV_ * DH_ * 2);
  float*  comp   = (float*) alloc((size_t)512 * 1024 * 4);

  auto blks = [](int n) { return (n + 255) / 256; };

  // pack inputs to f16 (weights transposed for contiguous WMMA B-operands)
  cvt_f32_f16  <<<blks(2048 * 1024), 256, 0, stream>>>(xh, x, 2048 * 1024);
  cvt_transpose<<<blks(1024 * 1024), 256, 0, stream>>>(wqt, Wq, 1024, 1024);
  cvt_transpose<<<blks(1024 * 2048), 256, 0, stream>>>(wkvt, Wkv, 1024, 2048);
  cvt_transpose<<<blks(1024 * 1024), 256, 0, stream>>>(woutt, Wout, 1024, 1024);
  cvt_f32_f16  <<<blks(2 * 1024 * 1024), 256, 0, stream>>>(convAh, mem, 2 * 1024 * 1024);
  cvt_f32_f16  <<<blks(H_ * KV_ * DH_), 256, 0, stream>>>(posh, pos, H_ * KV_ * DH_);
  build_kvin   <<<blks(B_ * KV_ * D_), 256, 0, stream>>>(kvinh, cmem, mem, x);
  build_convWT <<<blks(1024 * 4096), 256, 0, stream>>>(convWT, conv_w);

  // q = SCALE * x@Wq
  gemm_wmma<<<dim3(1024 / 128, 2048 / 64), 256, 0, stream>>>(
      xh, wqt, 2048, 1024, 1024, 1024, 1024, 1024, SCALE_, nullptr, nullptr, qh);
  // kv = kv_input@Wkv   (ld 2048: k = cols 0..1023, v = cols 1024..2047)
  gemm_wmma<<<dim3(2048 / 128, 6144 / 64), 256, 0, stream>>>(
      kvinh, wkvt, 6144, 2048, 1024, 1024, 1024, 2048, 1.0f, nullptr, nullptr, kvh);

  // attention: 2048 wave-tiles, 2 per block
  attn_kernel<<<1024, 64, 0, stream>>>(qh, kvh, posh, outh);

  // logits = out@Wout + b_out  -> d_out[0:2M]
  gemm_wmma<<<dim3(1024 / 128, 2048 / 64), 256, 0, stream>>>(
      outh, woutt, 2048, 1024, 1024, 1024, 1024, 1024, 1.0f, b_out, out, nullptr);

  // compressed = mem(512x4096) @ W2 + conv_b -> ws
  gemm_wmma<<<dim3(1024 / 128, 512 / 64), 256, 0, stream>>>(
      convAh, convWT, 512, 1024, 4096, 4096, 4096, 1024, 1.0f, conv_b, comp, nullptr);

  // new_mem = x ; new_cmem = [cmem[:,256:], compressed] ; aux = 0
  copy_f32  <<<blks(2 * 1024 * 1024), 256, 0, stream>>>(out + 2097152, x, 2 * 1024 * 1024);
  build_cmem<<<blks(2 * 1024 * 1024), 256, 0, stream>>>(out + 4194304, cmem, comp);
  write_aux <<<1, 32, 0, stream>>>(out + 6291456);
}
